// SimpleMTP_83468394430848
// MI455X (gfx1250) — matrix operand order repaired and checked by reading, not compiled, not run on hardware
//
#include <hip/hip_runtime.h>

// ---------------- problem constants ----------------
#define D_MODEL 1024
#define D_FF    2048
#define VOCABN  32000
#define BATCH   4
#define SEQT    512
#define NHEADS  3
#define NPOS    (SEQT - NHEADS)     // 509
#define NROWS   (BATCH * NPOS)      // 2036

typedef _Float16 half_t;
typedef __attribute__((ext_vector_type(16))) _Float16 v16h;
typedef __attribute__((ext_vector_type(8)))  float    v8f;

// ---------------- elementwise / conversion kernels ----------------

__global__ void cvt_f32_f16(const float* __restrict__ in, half_t* __restrict__ out, size_t n) {
    size_t i = (size_t)blockIdx.x * blockDim.x + threadIdx.x;
    size_t stride = (size_t)gridDim.x * blockDim.x;
    for (; i < n; i += stride) out[i] = (half_t)in[i];
}

__global__ void gather_rows(const float* __restrict__ embed, const int* __restrict__ ids,
                            float* __restrict__ out) {
    int r = blockIdx.x;
    int b = r / NPOS, i = r - b * NPOS;
    int t = ids[b * SEQT + i];
    const float* src = embed + (size_t)t * D_MODEL;
    float* dst = out + (size_t)r * D_MODEL;
    for (int j = threadIdx.x; j < D_MODEL; j += blockDim.x) dst[j] = src[j];
}

// merged[r, 0:1024]   = rmsnorm(h_prev[r])            (f16)
// merged[r,1024:2048] = rmsnorm(embed[ids[b,i+k+1]])  (f16)
__global__ void merge_rmsnorm(const float* __restrict__ hprev, const float* __restrict__ embed,
                              const int* __restrict__ ids, int head,
                              half_t* __restrict__ merged) {
    int r = blockIdx.x, tid = threadIdx.x;
    int b = r / NPOS, i = r - b * NPOS;
    __shared__ float red[256];
    for (int part = 0; part < 2; ++part) {
        const float* src;
        if (part == 0) {
            src = hprev + (size_t)r * D_MODEL;
        } else {
            int t = ids[b * SEQT + i + head + 1];
            src = embed + (size_t)t * D_MODEL;
        }
        float ss = 0.f;
        for (int j = tid; j < D_MODEL; j += 256) { float v = src[j]; ss += v * v; }
        red[tid] = ss; __syncthreads();
        for (int s = 128; s > 0; s >>= 1) { if (tid < s) red[tid] += red[tid + s]; __syncthreads(); }
        float inv = 1.0f / sqrtf(red[0] * (1.0f / D_MODEL) + 1e-8f);
        half_t* dst = merged + (size_t)r * (2 * D_MODEL) + part * D_MODEL;
        for (int j = tid; j < D_MODEL; j += 256) dst[j] = (half_t)(src[j] * inv);
        __syncthreads();
    }
}

// out = LayerNorm(x + y) * g + b   -> f32 (outF) and f16 (outB)
__global__ void add_layernorm(const float* __restrict__ x, const float* __restrict__ y,
                              const float* __restrict__ g, const float* __restrict__ bta,
                              float* __restrict__ outF, half_t* __restrict__ outB) {
    int r = blockIdx.x, tid = threadIdx.x;
    __shared__ float srow[D_MODEL];
    __shared__ float red[256];
    const float* xr = x + (size_t)r * D_MODEL;
    const float* yr = y + (size_t)r * D_MODEL;
    float sum = 0.f;
    for (int j = tid; j < D_MODEL; j += 256) { float v = xr[j] + yr[j]; srow[j] = v; sum += v; }
    red[tid] = sum; __syncthreads();
    for (int s = 128; s > 0; s >>= 1) { if (tid < s) red[tid] += red[tid + s]; __syncthreads(); }
    float mean = red[0] * (1.0f / D_MODEL);
    __syncthreads();
    float ss = 0.f;
    for (int j = tid; j < D_MODEL; j += 256) { float d = srow[j] - mean; ss += d * d; }
    red[tid] = ss; __syncthreads();
    for (int s = 128; s > 0; s >>= 1) { if (tid < s) red[tid] += red[tid + s]; __syncthreads(); }
    float inv = 1.0f / sqrtf(red[0] * (1.0f / D_MODEL) + 1e-5f);
    for (int j = tid; j < D_MODEL; j += 256) {
        float o = (srow[j] - mean) * inv * g[j] + bta[j];
        size_t idx = (size_t)r * D_MODEL + j;
        outF[idx] = o;
        if (outB) outB[idx] = (half_t)o;
    }
}

// ---------------- async copy helpers (CDNA5 ASYNCcnt path) ----------------

// 16B global -> LDS, tracked by ASYNCcnt (GLOBAL_LOAD_ASYNC_TO_LDS_B128, GV mode).
__device__ __forceinline__ void async_ld16(unsigned lds_off, const half_t* gp) {
    asm volatile("global_load_async_to_lds_b128 %0, %1, off"
                 :: "v"(lds_off), "v"(gp)
                 : "memory");
}
__device__ __forceinline__ void wait_async4() {   // leave newest 4 in flight
    asm volatile("s_wait_asynccnt 0x4" ::: "memory");
}
__device__ __forceinline__ void wait_async0() {   // drain
    asm volatile("s_wait_asynccnt 0x0" ::: "memory");
}

// ---------------- WMMA GEMM:  C[M,N] = A[M,K] * W[N,K]^T (+bias)(+relu) ----------------
// Block: 256 threads = 8 waves.  Block tile 128(M) x 128(N).
// Waves: 4 in M (32 rows) x 2 in N (64 cols); each wave = 2x4 WMMA 16x16 tiles.
// K staged in chunks of 32 through double-buffered LDS via async-to-LDS copies.

#define TILE_M 128
#define TILE_N 128
#define LDS_STRIDE 40   // 40 halfs = 80B row pitch (16B-aligned, bank-spreading)

__global__ __launch_bounds__(256) void gemm_f16_wmma(
    const half_t* __restrict__ A,    // row-major M x K
    const half_t* __restrict__ Wt,   // row-major N x K  (i.e. W, used as B^T)
    const float*  __restrict__ bias, // length N or null
    float*        __restrict__ outF, int ldF,    // optional f32 output
    half_t*       __restrict__ outB,             // optional f16 output (ld = N)
    int M, int N, int K, int relu) {

    __shared__ half_t As[2][TILE_M * LDS_STRIDE];
    __shared__ half_t Bs[2][TILE_N * LDS_STRIDE];

    const int tid   = threadIdx.x;
    const int lane  = tid & 31;
    const int wave  = tid >> 5;
    const int wm    = wave & 3;      // wave row (0..3) -> 32 rows
    const int wn    = wave >> 2;     // wave col (0..1) -> 64 cols
    const int lmod  = lane & 15;
    const int lhalf = lane >> 4;

    const int m0 = blockIdx.x * TILE_M;
    const int n0 = blockIdx.y * TILE_N;

    // staging: thread -> (row r, 16-elem col chunk h) for both A and B tiles
    const int r = tid >> 1;              // 0..127
    const int h = (tid & 1) * 16;        // 0 / 16

    int am = m0 + r; if (am >= M) am = M - 1;   // clamp: keeps async counts uniform
    const size_t a_row = (size_t)am * K;
    const size_t b_row = (size_t)(n0 + r) * K;  // N is a multiple of 128

    const unsigned lA[2] = { (unsigned)(size_t)(&As[0][0] + r * LDS_STRIDE + h),
                             (unsigned)(size_t)(&As[1][0] + r * LDS_STRIDE + h) };
    const unsigned lB[2] = { (unsigned)(size_t)(&Bs[0][0] + r * LDS_STRIDE + h),
                             (unsigned)(size_t)(&Bs[1][0] + r * LDS_STRIDE + h) };

    auto issue = [&](int stage, int k0) {
        const half_t* ga = A  + a_row + k0 + h;
        const half_t* gb = Wt + b_row + k0 + h;
        async_ld16(lA[stage],      ga);
        async_ld16(lA[stage] + 16, ga + 8);
        async_ld16(lB[stage],      gb);
        async_ld16(lB[stage] + 16, gb + 8);
    };

    v8f acc[2][4] = {};
    const int nk = K >> 5;   // K / 32, K is always a multiple of 32

    issue(0, 0);
    for (int kc = 0; kc < nk; ++kc) {
        const int  cur  = kc & 1;
        const bool more = (kc + 1) < nk;
        if (more) issue(cur ^ 1, (kc + 1) << 5);   // prefetch next stage (other buffer)
        if (more) wait_async4(); else wait_async0();// this wave's current stage landed
        __syncthreads();                            // all waves' stage data visible

        // ---- fragments in contiguous k order: element s of lane half h holds k = 16*h + s (A and B alike) ----
        v16h afrag[2], bfrag[4];
#pragma unroll
        for (int tm = 0; tm < 2; ++tm) {
            const half_t* base = &As[cur][0] + (wm * 32 + tm * 16 + lmod) * LDS_STRIDE;
            union { uint4 u[2]; v16h v; } f;
            f.u[0] = *(const uint4*)(base + lhalf * 16);
            f.u[1] = *(const uint4*)(base + lhalf * 16 + 8);
            afrag[tm] = f.v;
        }
#pragma unroll
        for (int tn = 0; tn < 4; ++tn) {
            const half_t* base = &Bs[cur][0] + (wn * 64 + tn * 16 + lmod) * LDS_STRIDE + lhalf * 16;
            union { uint4 u[2]; v16h v; } f;
            f.u[0] = *(const uint4*)(base);
            f.u[1] = *(const uint4*)(base + 8);
            bfrag[tn] = f.v;
        }

#pragma unroll
        for (int tm = 0; tm < 2; ++tm)
#pragma unroll
            for (int tn = 0; tn < 4; ++tn)
                acc[tm][tn] = __builtin_amdgcn_wmma_f32_16x16x32_f16(
                    false, afrag[tm], false, bfrag[tn],
                    (short)0, acc[tm][tn], false, false);

        __syncthreads();   // compute done before next iter's issue overwrites this buffer
    }

    // ---- epilogue: C layout — VGPR r: M = r + 8*lhalf, N = lmod ----
    const int mg = m0 + wm * 32 + 8 * lhalf;
    const int ng = n0 + wn * 64 + lmod;
#pragma unroll
    for (int tm = 0; tm < 2; ++tm) {
#pragma unroll
        for (int tn = 0; tn < 4; ++tn) {
            const int ncol = ng + tn * 16;
            const float bv = bias ? bias[ncol] : 0.f;
#pragma unroll
            for (int rr = 0; rr < 8; ++rr) {
                const int m = mg + tm * 16 + rr;
                if (m < M) {
                    float cv = acc[tm][tn][rr] + bv;
                    if (relu) cv = fmaxf(cv, 0.f);
                    if (outF) outF[(size_t)m * ldF + ncol] = cv;
                    if (outB) outB[(size_t)m * N + ncol] = (half_t)cv;
                }
            }
        }
    }
}

// ---------------- host orchestration ----------------

extern "C" void kernel_launch(void* const* d_in, const int* in_sizes, int n_in,
                              void* d_out, int out_size, void* d_ws, size_t ws_size,
                              hipStream_t stream) {
    (void)in_sizes; (void)n_in; (void)out_size; (void)ws_size;

    const int*   token_ids = (const int*)d_in[0];
    const float* embed     = (const float*)d_in[1];
    // per-layer params in dict order:
    // 0 Wmerge 1 bmerge 2 Wv 3 bv 4 Wo 5 bo 6 W1 7 b1 8 W2 9 b2 10 ln1_g 11 ln1_b 12 ln2_g 13 ln2_b
    const float* P[NHEADS][14];
    for (int k = 0; k < NHEADS; ++k)
        for (int j = 0; j < 14; ++j)
            P[k][j] = (const float*)d_in[2 + k * 14 + j];

    // ---- carve workspace ----
    char* wp = (char*)d_ws;
    auto carve = [&](size_t bytes) -> void* {
        void* p = (void*)wp;
        wp += (bytes + 255) & ~(size_t)255;
        return p;
    };
    half_t* embed_h = (half_t*)carve((size_t)VOCABN * D_MODEL * sizeof(half_t));
    half_t *Wm_h[NHEADS], *Wv_h[NHEADS], *Wo_h[NHEADS], *W1_h[NHEADS], *W2_h[NHEADS];
    for (int k = 0; k < NHEADS; ++k) {
        Wm_h[k] = (half_t*)carve((size_t)D_MODEL * (2 * D_MODEL) * sizeof(half_t));
        Wv_h[k] = (half_t*)carve((size_t)D_MODEL * D_MODEL * sizeof(half_t));
        Wo_h[k] = (half_t*)carve((size_t)D_MODEL * D_MODEL * sizeof(half_t));
        W1_h[k] = (half_t*)carve((size_t)D_FF * D_MODEL * sizeof(half_t));
        W2_h[k] = (half_t*)carve((size_t)D_MODEL * D_FF * sizeof(half_t));
    }
    float*  hbuf0  = (float*)carve((size_t)NROWS * D_MODEL * sizeof(float));
    float*  hbuf1  = (float*)carve((size_t)NROWS * D_MODEL * sizeof(float));
    half_t* merged = (half_t*)carve((size_t)NROWS * (2 * D_MODEL) * sizeof(half_t));
    float*  xf     = (float*)carve((size_t)NROWS * D_MODEL * sizeof(float));
    half_t* xb     = (half_t*)carve((size_t)NROWS * D_MODEL * sizeof(half_t));
    half_t* vb     = (half_t*)carve((size_t)NROWS * D_MODEL * sizeof(half_t));
    float*  attn   = (float*)carve((size_t)NROWS * D_MODEL * sizeof(float));
    float*  x2f    = (float*)carve((size_t)NROWS * D_MODEL * sizeof(float));
    half_t* x2b    = (half_t*)carve((size_t)NROWS * D_MODEL * sizeof(half_t));
    half_t* ffm    = (half_t*)carve((size_t)NROWS * D_FF * sizeof(half_t));
    float*  fff    = (float*)carve((size_t)NROWS * D_MODEL * sizeof(float));
    half_t* hcb    = (half_t*)carve((size_t)NROWS * D_MODEL * sizeof(half_t));
    float*  hbuf[2] = { hbuf0, hbuf1 };

    // ---- one-time (per call) f32 -> f16 conversions ----
    cvt_f32_f16<<<4096, 256, 0, stream>>>(embed, embed_h, (size_t)VOCABN * D_MODEL);
    for (int k = 0; k < NHEADS; ++k) {
        cvt_f32_f16<<<2048, 256, 0, stream>>>(P[k][0], Wm_h[k], (size_t)D_MODEL * (2 * D_MODEL));
        cvt_f32_f16<<<2048, 256, 0, stream>>>(P[k][2], Wv_h[k], (size_t)D_MODEL * D_MODEL);
        cvt_f32_f16<<<2048, 256, 0, stream>>>(P[k][4], Wo_h[k], (size_t)D_MODEL * D_MODEL);
        cvt_f32_f16<<<2048, 256, 0, stream>>>(P[k][6], W1_h[k], (size_t)D_FF * D_MODEL);
        cvt_f32_f16<<<2048, 256, 0, stream>>>(P[k][8], W2_h[k], (size_t)D_MODEL * D_FF);
    }

    // h_prev init = embed[token_ids[:, 0:509]]
    gather_rows<<<NROWS, 256, 0, stream>>>(embed, token_ids, hbuf[0]);

    dim3 blk(256);
    auto G = [](int M, int N) { return dim3((unsigned)((M + TILE_M - 1) / TILE_M),
                                            (unsigned)(N / TILE_N)); };

    for (int k = 0; k < NHEADS; ++k) {
        float* hprev = hbuf[k & 1];
        float* hcur  = hbuf[(k & 1) ^ 1];

        merge_rmsnorm<<<NROWS, 256, 0, stream>>>(hprev, embed, token_ids, k, merged);

        // x = merged @ Wmerge^T + bmerge          (f32 + f16 out)
        gemm_f16_wmma<<<G(NROWS, D_MODEL), blk, 0, stream>>>(
            merged, Wm_h[k], P[k][1], xf, D_MODEL, xb, NROWS, D_MODEL, 2 * D_MODEL, 0);
        // v = x @ Wv^T + bv                       (f16 out only)
        gemm_f16_wmma<<<G(NROWS, D_MODEL), blk, 0, stream>>>(
            xb, Wv_h[k], P[k][3], nullptr, 0, vb, NROWS, D_MODEL, D_MODEL, 0);
        // attn = v @ Wo^T + bo                    (f32 out only)
        gemm_f16_wmma<<<G(NROWS, D_MODEL), blk, 0, stream>>>(
            vb, Wo_h[k], P[k][5], attn, D_MODEL, nullptr, NROWS, D_MODEL, D_MODEL, 0);
        // x2 = LN(x + attn)
        add_layernorm<<<NROWS, 256, 0, stream>>>(xf, attn, P[k][10], P[k][11], x2f, x2b);
        // ffm = relu(x2 @ W1^T + b1)              (f16 out only, fused relu)
        gemm_f16_wmma<<<G(NROWS, D_FF), blk, 0, stream>>>(
            x2b, W1_h[k], P[k][7], nullptr, 0, ffm, NROWS, D_FF, D_MODEL, 1);
        // ff = ffm @ W2^T + b2                    (f32 out only)
        gemm_f16_wmma<<<G(NROWS, D_MODEL), blk, 0, stream>>>(
            ffm, W2_h[k], P[k][9], fff, D_MODEL, nullptr, NROWS, D_MODEL, D_FF, 0);
        // h_curr = LN(x2 + ff)
        add_layernorm<<<NROWS, 256, 0, stream>>>(x2f, fff, P[k][12], P[k][13], hcur, hcb);
        // logits = h_curr @ embed^T -> d_out[:, :, k, :]  (ld = 3*VOCAB, col offset k*VOCAB)
        gemm_f16_wmma<<<G(NROWS, VOCABN), blk, 0, stream>>>(
            hcb, embed_h, nullptr,
            (float*)d_out + (size_t)k * VOCABN, NHEADS * VOCABN, nullptr,
            NROWS, VOCABN, D_MODEL, 0);
    }
}
